// AttentionDecoder_86466281603704
// MI455X (gfx1250) — matrix operand order repaired and checked by reading, not compiled, not run on hardware
//
#include <hip/hip_runtime.h>
#include <hip/hip_bf16.h>

// ---------------------------------------------------------------------------
// Problem constants (from reference)
// ---------------------------------------------------------------------------
#define LAY 2
#define BB  2
#define TT  2048
#define CC  512
#define HH  8
#define FFN 1024
#define HS  64
#define MM  (BB * TT)          // 4096 rows
#define EPS_RMS 1.1920929e-07f

typedef __attribute__((ext_vector_type(16))) _Float16 v16h;
typedef __attribute__((ext_vector_type(8)))  float    v8f;

// ---------------------------------------------------------------------------
// WMMA wrapper: D = A(16x32 f16) * B(32x16 f16) + C(16x16 f32)
// ---------------------------------------------------------------------------
__device__ __forceinline__ v8f wmma16(v16h a, v16h b, v8f c) {
  return __builtin_amdgcn_wmma_f32_16x16x32_f16(false, a, false, b,
                                                (short)0, c, false, false);
}

// ---------------------------------------------------------------------------
// Async global -> LDS copy, 16 bytes per lane (gfx1250 ASYNCcnt path).
// ---------------------------------------------------------------------------
__device__ __forceinline__ void async_b128(unsigned lds, const void* g) {
  asm volatile("global_load_async_to_lds_b128 %0, %1, off"
               :: "v"(lds), "v"((unsigned long long)g)
               : "memory");
}
__device__ __forceinline__ void wait_async0() {
  asm volatile("s_wait_asynccnt 0x0" ::: "memory");
}
__device__ __forceinline__ unsigned lds_addr(const void* p) {
  return (unsigned)(unsigned long long)p;
}

// ---------------------------------------------------------------------------
// Fragment loaders (wave32 WMMA operands, contiguous k order shared by A and B), all via 16B LDS reads.
// A matrix 16x32 f16, source row-major [rows][ldk] (ldk multiple of 8):
//   lane half h: K = 16h .. 16h+15 (fully contiguous).
// ---------------------------------------------------------------------------
__device__ __forceinline__ v16h frag_a16(const _Float16* p, int ldk,
                                         int row0, int k0) {
  int lane = threadIdx.x & 31;
  int half = lane >> 4;
  const _Float16* r = p + (size_t)(row0 + (lane & 15)) * ldk + k0 + 16 * half;
  union { v16h v; uint4 q[2]; } u;
  u.q[0] = *(const uint4*)r;
  u.q[1] = *(const uint4*)(r + 8);
  return u.v;
}

// B matrix 32x16 f16, source stored [N][ldk] (N-major, K contiguous):
//   lane half h, N = n0 + (lane&15): K = 16h .. 16h+15 (fully contiguous).
__device__ __forceinline__ v16h frag_bT(const _Float16* p, int ldk,
                                        int n0, int k0) {
  int lane = threadIdx.x & 31;
  int half = lane >> 4;
  const _Float16* r = p + (size_t)(n0 + (lane & 15)) * ldk + k0 + 16 * half;
  union { v16h v; uint4 q[2]; } u;
  u.q[0] = *(const uint4*)r;
  u.q[1] = *(const uint4*)(r + 8);
  return u.v;
}

// ---------------------------------------------------------------------------
// Elementwise helpers
// ---------------------------------------------------------------------------
__global__ __launch_bounds__(256) void copy_f32(const float* __restrict__ src,
                                                float* __restrict__ dst, int n) {
  int i = blockIdx.x * 256 + threadIdx.x;
  if (i < n) dst[i] = src[i];
}

// Transpose + convert: src f32 [R][Cn] -> dst f16 [Cn][R]  (weight -> [N][K])
__global__ __launch_bounds__(256) void convT_f16(const float* __restrict__ src,
                                                 _Float16* __restrict__ dst,
                                                 int R, int Cn) {
  int idx = blockIdx.x * 256 + threadIdx.x;
  if (idx >= R * Cn) return;
  int c = idx / R;   // output row (N)
  int r = idx % R;   // output col (K)
  dst[idx] = (_Float16)src[(size_t)r * Cn + c];
}

// Per-head weights [H][C][HS] f32 -> transposed packed [H*HS][C] f16 ([N][K])
__global__ __launch_bounds__(256) void pack_heads_T(const float* __restrict__ w,
                                                    _Float16* __restrict__ out) {
  int idx = blockIdx.x * 256 + threadIdx.x;   // 0 .. C*C-1
  if (idx >= CC * CC) return;
  int n = idx / CC;           // h*HS + d
  int k = idx % CC;           // c
  int h = n >> 6;
  int d = n & 63;
  out[idx] = (_Float16)w[((size_t)h * CC + k) * HS + d];
}

// RMSNorm: one wave per row of [M, C] fp32 -> f16
__global__ __launch_bounds__(256) void rmsnorm_f16(const float* __restrict__ x,
                                                   const float* __restrict__ g,
                                                   _Float16* __restrict__ out) {
  int wave = threadIdx.x >> 5;
  int lane = threadIdx.x & 31;
  int row  = blockIdx.x * 8 + wave;
  const float* xr = x + (size_t)row * CC;
  float vals[CC / 32];
  float ss = 0.0f;
#pragma unroll
  for (int i = 0; i < CC / 32; ++i) {
    float v = xr[lane + i * 32];
    vals[i] = v;
    ss += v * v;
  }
#pragma unroll
  for (int m = 16; m >= 1; m >>= 1) ss += __shfl_xor(ss, m, 32);
  float rs = rsqrtf(ss * (1.0f / CC) + EPS_RMS);
  _Float16* orow = out + (size_t)row * CC;
#pragma unroll
  for (int i = 0; i < CC / 32; ++i)
    orow[lane + i * 32] = (_Float16)(vals[i] * rs * g[lane + i * 32]);
}

// ---------------------------------------------------------------------------
// Tiled WMMA GEMM: out[M,N] = epi(A[M,K]f16 * Wt[N,K]f16 + bias)
//   Wt is PRE-TRANSPOSED ([N][K]) so B fragments are b128 loads.
//   EPI 0: store f16                EPI 1: exact-GELU, store f16
//   EPI 2: store f32 = res+acc+b    EPI 3: store f16 scaled by 1/8 (Q proj)
// 256 threads = 8 waves (4x2); tile 128x128, BK=32.
// DOUBLE-BUFFERED async global->LDS staging: DMA of tile k+1 overlaps the
// WMMAs of tile k; single barrier per K-step.
// ---------------------------------------------------------------------------
#define GBM 128
#define GBN 128
#define GBK 32
#define GLDA (GBK + 8)   // 40 f16 = 80B, 16B-aligned stride

template <int EPI>
__global__ __launch_bounds__(256) void gemm_f16(
    const _Float16* __restrict__ A, const _Float16* __restrict__ Wt,
    const float* __restrict__ bias, const float* __restrict__ resid,
    void* __restrict__ outp, int M, int N, int K) {
  __shared__ _Float16 As[2][GBM][GLDA];
  __shared__ _Float16 Bs[2][GBN][GLDA];

  int m0 = blockIdx.y * GBM;
  int n0 = blockIdx.x * GBN;
  int wave = threadIdx.x >> 5;
  int wr = wave & 3;   // row subtile (32 rows)
  int wc = wave >> 2;  // col subtile (64 cols)

  // per-thread staging addresses: 2 chunks for A, 2 for B (16B each)
  int rs = threadIdx.x >> 2;        // 0..63
  int cs = (threadIdx.x & 3) * 8;   // 0,8,16,24
  const _Float16* gA0 = A  + (size_t)(m0 + rs) * K + cs;
  const _Float16* gA1 = A  + (size_t)(m0 + rs + 64) * K + cs;
  const _Float16* gB0 = Wt + (size_t)(n0 + rs) * K + cs;
  const _Float16* gB1 = Wt + (size_t)(n0 + rs + 64) * K + cs;
  unsigned sA0[2], sA1[2], sB0[2], sB1[2];
#pragma unroll
  for (int bf = 0; bf < 2; ++bf) {
    sA0[bf] = lds_addr(&As[bf][rs][cs]);
    sA1[bf] = lds_addr(&As[bf][rs + 64][cs]);
    sB0[bf] = lds_addr(&Bs[bf][rs][cs]);
    sB1[bf] = lds_addr(&Bs[bf][rs + 64][cs]);
  }

  v8f acc[2][4];
#pragma unroll
  for (int i = 0; i < 2; ++i)
#pragma unroll
    for (int j = 0; j < 4; ++j) acc[i][j] = (v8f){0,0,0,0,0,0,0,0};

  // prologue: stage tile 0 into buffer 0
  async_b128(sA0[0], gA0);
  async_b128(sA1[0], gA1);
  async_b128(sB0[0], gB0);
  async_b128(sB1[0], gB1);
  gA0 += GBK; gA1 += GBK; gB0 += GBK; gB1 += GBK;

  int nsteps = K / GBK;
  for (int it = 0; it < nsteps; ++it) {
    int buf = it & 1;
    wait_async0();        // tile `it` landed in LDS (this wave's copies)
    __syncthreads();      // ...and everyone else's; alt buffer now reusable
    if (it + 1 < nsteps) {
      int nb = buf ^ 1;   // prefetch tile it+1 (overlaps WMMAs below)
      async_b128(sA0[nb], gA0);
      async_b128(sA1[nb], gA1);
      async_b128(sB0[nb], gB0);
      async_b128(sB1[nb], gB1);
      gA0 += GBK; gA1 += GBK; gB0 += GBK; gB1 += GBK;
    }
    v16h af0 = frag_a16(&As[buf][0][0], GLDA, wr * 32, 0);
    v16h af1 = frag_a16(&As[buf][0][0], GLDA, wr * 32 + 16, 0);
#pragma unroll
    for (int j = 0; j < 4; ++j) {
      v16h bf = frag_bT(&Bs[buf][0][0], GLDA, wc * 64 + j * 16, 0);
      acc[0][j] = wmma16(af0, bf, acc[0][j]);
      acc[1][j] = wmma16(af1, bf, acc[1][j]);
    }
  }

  int lane = threadIdx.x & 31;
  int half = lane >> 4;
  int nlane = lane & 15;
#pragma unroll
  for (int i2 = 0; i2 < 2; ++i2) {
#pragma unroll
    for (int j = 0; j < 4; ++j) {
      int col = n0 + wc * 64 + j * 16 + nlane;
      float bv = bias ? bias[col] : 0.0f;
#pragma unroll
      for (int i = 0; i < 8; ++i) {
        int row = m0 + wr * 32 + i2 * 16 + 8 * half + i;
        float x = acc[i2][j][i] + bv;
        if (EPI == 1) x = 0.5f * x * (1.0f + erff(x * 0.70710678118654752f));
        if (EPI == 3) x *= 0.125f;   // fold HS^-0.5 into Q projection
        if (EPI == 2) {
          ((float*)outp)[(size_t)row * N + col] =
              resid[(size_t)row * N + col] + x;
        } else {
          ((_Float16*)outp)[(size_t)row * N + col] = (_Float16)x;
        }
      }
    }
  }
}

// ---------------------------------------------------------------------------
// Flash attention. Q,K,V,O layouts: [B, T, H*HS] f16 (head at column h*HS).
// Q is pre-scaled by HS^-0.5 (GEMM epilogue 3).
// 128 threads = 4 waves; each wave owns 16 query rows (64 per block).
// DOUBLE-BUFFERED KV tiles: K via async global->LDS, V via load+transpose
// scatter, both prefetched for tile k+1 while tile k computes. Ps is
// wave-private (in-order LDS within a wave) -> ONE barrier per KV tile.
// ---------------------------------------------------------------------------
#define ALD (HS + 8)     // 72 f16 = 144B stride, 16B-aligned
#define PLD (64 + 8)

template <bool CAUSAL>
__global__ __launch_bounds__(128) void flash_attn(
    const _Float16* __restrict__ Q, const _Float16* __restrict__ Kv,
    const _Float16* __restrict__ Vv, _Float16* __restrict__ O,
    int Tq, int Tk) {
  __shared__ _Float16 Qs[64][ALD];
  __shared__ _Float16 Ks[2][64][ALD];  // [kv][d] == [N][K] for S = Q*K^T
  __shared__ _Float16 Vt[2][HS][PLD];  // [d][kv] == [N][K] for O = P*V
  __shared__ _Float16 Ps[4][16][PLD];  // per-wave P tile, [m][kv]

  int b = blockIdx.y / HH;
  int h = blockIdx.y % HH;
  int q0 = blockIdx.x * 64;

  const _Float16* Qg = Q + (size_t)b * Tq * CC + h * HS;
  const _Float16* Kg = Kv + (size_t)b * Tk * CC + h * HS;
  const _Float16* Vg = Vv + (size_t)b * Tk * CC + h * HS;

  // per-thread staging geometry: 4 rows (r, r+16, r+32, r+48), one 16B chunk
  int rs = threadIdx.x >> 3;        // 0..15
  int cs = (threadIdx.x & 7) * 8;   // d base

  // stage Q tile (64 x 64) via async copies
#pragma unroll
  for (int u = 0; u < 4; ++u)
    async_b128(lds_addr(&Qs[rs + 16 * u][cs]),
               Qg + (size_t)(q0 + rs + 16 * u) * CC + cs);

  int kend = CAUSAL ? (q0 + 64) : Tk;
  int nsteps = kend / 64;

  const _Float16* Kp = Kg + (size_t)rs * CC + cs;
  const _Float16* Vp = Vg + (size_t)rs * CC + cs;
  unsigned sK[2][4];
#pragma unroll
  for (int bf = 0; bf < 2; ++bf)
#pragma unroll
    for (int u = 0; u < 4; ++u) sK[bf][u] = lds_addr(&Ks[bf][rs + 16 * u][cs]);

  // prologue: stage KV tile 0 into buffer 0 (K async, V transposed scatter)
#pragma unroll
  for (int u = 0; u < 4; ++u)
    async_b128(sK[0][u], Kp + (size_t)(16 * u) * CC);
#pragma unroll
  for (int u = 0; u < 4; ++u) {
    uint4 w = *(const uint4*)(Vp + (size_t)(16 * u) * CC);
    const _Float16* e = (const _Float16*)&w;
#pragma unroll
    for (int q = 0; q < 8; ++q) Vt[0][cs + q][rs + 16 * u] = e[q];
  }
  Kp += (size_t)64 * CC;
  Vp += (size_t)64 * CC;

  wait_async0();     // Q + K tile 0 landed
  __syncthreads();

  int wave = threadIdx.x >> 5;
  int lane = threadIdx.x & 31;
  int half = lane >> 4;
  int nlane = lane & 15;

  v16h qf0 = frag_a16(&Qs[0][0], ALD, wave * 16, 0);
  v16h qf1 = frag_a16(&Qs[0][0], ALD, wave * 16, 32);

  v8f o[4];
#pragma unroll
  for (int j = 0; j < 4; ++j) o[j] = (v8f){0,0,0,0,0,0,0,0};
  float mrow[8], lrow[8];
#pragma unroll
  for (int i = 0; i < 8; ++i) { mrow[i] = -3.0e38f; lrow[i] = 0.0f; }

  for (int it = 0; it < nsteps; ++it) {
    int buf = it & 1;
    if (it > 0) {
      wait_async0();     // K tile `it` landed
      __syncthreads();   // ...from all waves; alt KV buffer now reusable
    }
    if (it + 1 < nsteps) {
      int nb = buf ^ 1;  // prefetch KV tile it+1 (overlaps compute below)
#pragma unroll
      for (int u = 0; u < 4; ++u)
        async_b128(sK[nb][u], Kp + (size_t)(16 * u) * CC);
#pragma unroll
      for (int u = 0; u < 4; ++u) {
        uint4 w = *(const uint4*)(Vp + (size_t)(16 * u) * CC);
        const _Float16* e = (const _Float16*)&w;
#pragma unroll
        for (int q = 0; q < 8; ++q) Vt[nb][cs + q][rs + 16 * u] = e[q];
      }
      Kp += (size_t)64 * CC;
      Vp += (size_t)64 * CC;
    }

    // S = Q * K^T  (16 x 64 per wave, 4 kv subtiles)
    int k0 = it * 64;
    float s[4][8];
#pragma unroll
    for (int j = 0; j < 4; ++j) {
      v16h kf0 = frag_bT(&Ks[buf][0][0], ALD, j * 16, 0);
      v16h kf1 = frag_bT(&Ks[buf][0][0], ALD, j * 16, 32);
      v8f sa = (v8f){0,0,0,0,0,0,0,0};
      sa = wmma16(qf0, kf0, sa);
      sa = wmma16(qf1, kf1, sa);
#pragma unroll
      for (int i = 0; i < 8; ++i) s[j][i] = sa[i];
    }
    if (CAUSAL) {
#pragma unroll
      for (int j = 0; j < 4; ++j)
#pragma unroll
        for (int i = 0; i < 8; ++i) {
          int qr = q0 + wave * 16 + 8 * half + i;
          int kc = k0 + j * 16 + nlane;
          if (kc > qr) s[j][i] = -1.0e30f;
        }
    }
    // online softmax: row stats across 4 subtiles + 16 lanes of this half
    float scl[8];
#pragma unroll
    for (int i = 0; i < 8; ++i) {
      float mx = s[0][i];
#pragma unroll
      for (int j = 1; j < 4; ++j) mx = fmaxf(mx, s[j][i]);
#pragma unroll
      for (int m = 8; m >= 1; m >>= 1) mx = fmaxf(mx, __shfl_xor(mx, m, 32));
      float mnew = fmaxf(mrow[i], mx);
      scl[i] = __expf(mrow[i] - mnew);
      float rsum = 0.0f;
#pragma unroll
      for (int j = 0; j < 4; ++j) {
        s[j][i] = __expf(s[j][i] - mnew);
        rsum += s[j][i];
      }
#pragma unroll
      for (int m = 8; m >= 1; m >>= 1) rsum += __shfl_xor(rsum, m, 32);
      lrow[i] = lrow[i] * scl[i] + rsum;
      mrow[i] = mnew;
    }
#pragma unroll
    for (int j = 0; j < 4; ++j)
#pragma unroll
      for (int i = 0; i < 8; ++i) o[j][i] *= scl[i];

    // stage P (wave-private; LDS ops are in-order within a wave -> no barrier)
#pragma unroll
    for (int j = 0; j < 4; ++j)
#pragma unroll
      for (int i = 0; i < 8; ++i)
        Ps[wave][8 * half + i][j * 16 + nlane] = (_Float16)s[j][i];

    // O += P * V  (B-frags from transposed V tile: contiguous b128 reads)
    v16h pf0 = frag_a16(&Ps[wave][0][0], PLD, 0, 0);
    v16h pf1 = frag_a16(&Ps[wave][0][0], PLD, 0, 32);
#pragma unroll
    for (int j = 0; j < 4; ++j) {
      v16h vf0 = frag_bT(&Vt[buf][0][0], PLD, j * 16, 0);
      v16h vf1 = frag_bT(&Vt[buf][0][0], PLD, j * 16, 32);
      o[j] = wmma16(pf0, vf0, o[j]);
      o[j] = wmma16(pf1, vf1, o[j]);
    }
  }

  _Float16* Og = O + (size_t)b * Tq * CC + h * HS;
#pragma unroll
  for (int j = 0; j < 4; ++j)
#pragma unroll
    for (int i = 0; i < 8; ++i) {
      int r = q0 + wave * 16 + 8 * half + i;
      int c = j * 16 + nlane;
      Og[(size_t)r * CC + c] = (_Float16)(o[j][i] / lrow[i]);
    }
}

// ---------------------------------------------------------------------------
// Host orchestration
// ---------------------------------------------------------------------------
extern "C" void kernel_launch(void* const* d_in, const int* in_sizes, int n_in,
                              void* d_out, int out_size, void* d_ws, size_t ws_size,
                              hipStream_t stream) {
  (void)in_sizes; (void)n_in; (void)out_size; (void)ws_size;

  const float* hidden = (const float*)d_in[0];
  const float* target = (const float*)d_in[1];
  const float* Wq_s = (const float*)d_in[2];
  const float* Wk_s = (const float*)d_in[3];
  const float* Wv_s = (const float*)d_in[4];
  const float* Wo_s = (const float*)d_in[5];
  const float* bo_s = (const float*)d_in[6];
  const float* Wq_x = (const float*)d_in[7];
  const float* Wk_x = (const float*)d_in[8];
  const float* Wv_x = (const float*)d_in[9];
  const float* Wo_x = (const float*)d_in[10];
  const float* bo_x = (const float*)d_in[11];
  const float* W1  = (const float*)d_in[12];
  const float* b1  = (const float*)d_in[13];
  const float* W2  = (const float*)d_in[14];
  const float* b2  = (const float*)d_in[15];
  const float* g1  = (const float*)d_in[16];
  const float* g2  = (const float*)d_in[17];
  const float* g3  = (const float*)d_in[18];
  const float* g4  = (const float*)d_in[19];

  float* t = (float*)d_out;  // running residual stream [M, C] f32

  // workspace carve-up
  char* ws = (char*)d_ws;
  size_t off = 0;
  auto carve = [&](size_t bytes) {
    size_t o = off;
    off += (bytes + 255) & ~(size_t)255;
    return (void*)(ws + o);
  };
  _Float16* xn  = (_Float16*)carve((size_t)MM * CC * 2);
  _Float16* qb  = (_Float16*)carve((size_t)MM * CC * 2);
  _Float16* kb  = (_Float16*)carve((size_t)MM * CC * 2);
  _Float16* vb  = (_Float16*)carve((size_t)MM * CC * 2);
  _Float16* ao  = (_Float16*)carve((size_t)MM * CC * 2);
  _Float16* hff = (_Float16*)carve((size_t)MM * FFN * 2);
  _Float16* pwq = (_Float16*)carve((size_t)CC * CC * 2);   // [N][K]
  _Float16* pwk = (_Float16*)carve((size_t)CC * CC * 2);
  _Float16* pwv = (_Float16*)carve((size_t)CC * CC * 2);
  _Float16* pwo = (_Float16*)carve((size_t)CC * CC * 2);
  _Float16* pw1 = (_Float16*)carve((size_t)CC * FFN * 2);  // [FF][C]
  _Float16* pw2 = (_Float16*)carve((size_t)FFN * CC * 2);  // [C][FF]

  const dim3 blk256(256), blk128(128);
  const dim3 ln_grid(MM / 8);
  const dim3 gemm512(CC / GBN, MM / GBM);    // N=512
  const dim3 gemm1024(FFN / GBN, MM / GBM);  // N=1024
  const dim3 attn_grid(TT / 64, BB * HH);
  const dim3 pack_grid((CC * CC + 255) / 256);
  const dim3 ffw_grid((CC * FFN + 255) / 256);

  // t = target
  copy_f32<<<(MM * CC + 255) / 256, blk256, 0, stream>>>(target, t, MM * CC);

  for (int l = 0; l < LAY; ++l) {
    // ---------------- masked self-attention ----------------
    pack_heads_T<<<pack_grid, blk256, 0, stream>>>(Wq_s + (size_t)l * HH * CC * HS, pwq);
    pack_heads_T<<<pack_grid, blk256, 0, stream>>>(Wk_s + (size_t)l * HH * CC * HS, pwk);
    pack_heads_T<<<pack_grid, blk256, 0, stream>>>(Wv_s + (size_t)l * HH * CC * HS, pwv);
    convT_f16<<<pack_grid, blk256, 0, stream>>>(Wo_s + (size_t)l * CC * CC, pwo, CC, CC);

    rmsnorm_f16<<<ln_grid, blk256, 0, stream>>>(t, g1 + l * CC, xn);
    gemm_f16<3><<<gemm512, blk256, 0, stream>>>(xn, pwq, nullptr, nullptr, qb, MM, CC, CC);
    gemm_f16<0><<<gemm512, blk256, 0, stream>>>(xn, pwk, nullptr, nullptr, kb, MM, CC, CC);
    gemm_f16<0><<<gemm512, blk256, 0, stream>>>(xn, pwv, nullptr, nullptr, vb, MM, CC, CC);
    flash_attn<true><<<attn_grid, blk128, 0, stream>>>(qb, kb, vb, ao, TT, TT);
    gemm_f16<2><<<gemm512, blk256, 0, stream>>>(ao, pwo, bo_s + l * CC, t, t, MM, CC, CC);

    // ---------------- cross-attention ----------------
    pack_heads_T<<<pack_grid, blk256, 0, stream>>>(Wq_x + (size_t)l * HH * CC * HS, pwq);
    pack_heads_T<<<pack_grid, blk256, 0, stream>>>(Wk_x + (size_t)l * HH * CC * HS, pwk);
    pack_heads_T<<<pack_grid, blk256, 0, stream>>>(Wv_x + (size_t)l * HH * CC * HS, pwv);
    convT_f16<<<pack_grid, blk256, 0, stream>>>(Wo_x + (size_t)l * CC * CC, pwo, CC, CC);

    rmsnorm_f16<<<ln_grid, blk256, 0, stream>>>(hidden, g2 + l * CC, xn);
    gemm_f16<0><<<gemm512, blk256, 0, stream>>>(xn, pwk, nullptr, nullptr, kb, MM, CC, CC);
    gemm_f16<0><<<gemm512, blk256, 0, stream>>>(xn, pwv, nullptr, nullptr, vb, MM, CC, CC);
    rmsnorm_f16<<<ln_grid, blk256, 0, stream>>>(t, g3 + l * CC, xn);
    gemm_f16<3><<<gemm512, blk256, 0, stream>>>(xn, pwq, nullptr, nullptr, qb, MM, CC, CC);
    flash_attn<false><<<attn_grid, blk128, 0, stream>>>(qb, kb, vb, ao, TT, TT);
    gemm_f16<2><<<gemm512, blk256, 0, stream>>>(ao, pwo, bo_x + l * CC, t, t, MM, CC, CC);

    // ---------------- GELU FFN ----------------
    convT_f16<<<ffw_grid, blk256, 0, stream>>>(W1 + (size_t)l * CC * FFN, pw1, CC, FFN);
    convT_f16<<<ffw_grid, blk256, 0, stream>>>(W2 + (size_t)l * FFN * CC, pw2, FFN, CC);

    rmsnorm_f16<<<ln_grid, blk256, 0, stream>>>(t, g4 + l * CC, xn);
    gemm_f16<1><<<gemm1024, blk256, 0, stream>>>(xn, pw1, b1 + l * FFN, nullptr, hff, MM, FFN, CC);
    gemm_f16<2><<<gemm512, blk256, 0, stream>>>(hff, pw2, b2 + l * CC, t, t, MM, CC, FFN);
  }
}
